// ComplexMamba3Layer_7086696039113
// MI455X (gfx1250) — compile-verified, matrix operand order checked by reading, not run on hardware
//
#include <hip/hip_runtime.h>
#include <hip/hip_bf16.h>
#include <math.h>

// ---------------------------------------------------------------------------
// ComplexMamba3 layer for MI455X (gfx1250, wave32, WMMA, TDM).
// Pipeline: norm -> gate GEMM (wmma, fused sigmoid) -> rotate -> causal conv
// -> act -> dt proj (wave reduce) -> complex B-proj (dual wmma, fused dt scale)
// -> chunked parallel scan -> complex C-proj (dual wmma, fused f16 emit)
// -> complex out-proj (dual wmma, fused scale+residual+interleave).
// B tiles of every GEMM are staged in LDS via the Tensor Data Mover
// (tensor_load_to_lds + s_wait_tensorcnt), with a cooperative global->LDS
// fallback if the TDM builtin is unavailable.
// ---------------------------------------------------------------------------

typedef __attribute__((ext_vector_type(16))) _Float16 v16h;
typedef __attribute__((ext_vector_type(8)))  _Float16 v8h;
typedef __attribute__((ext_vector_type(8)))  float    v8f;
typedef __attribute__((ext_vector_type(4)))  unsigned int v4u;
typedef __attribute__((ext_vector_type(8)))  int          v8i;
typedef __attribute__((ext_vector_type(4)))  int          v4i;

#if defined(__has_builtin)
# if __has_builtin(__builtin_amdgcn_tensor_load_to_lds) && \
     __has_builtin(__builtin_amdgcn_s_wait_tensorcnt)
#  define CM3_HAS_TDM 1
# endif
#endif
#ifndef CM3_HAS_TDM
# define CM3_HAS_TDM 0
#endif

// Problem constants
constexpr int    Bn = 4, Sn = 4096, Dn = 1024, Gn = 8, Nn = 64;
constexpr int    BS = Bn * Sn;                       // 16384 tokens
constexpr size_t BSD  = (size_t)BS * Dn;             // 16,777,216
constexpr size_t BSG  = (size_t)BS * Gn;             // 131,072
constexpr size_t BSGN = (size_t)BS * Gn * Nn;        // 8,388,608

// ---------------------------------------------------------------------------
// Small elementwise helpers
// ---------------------------------------------------------------------------
__global__ __launch_bounds__(256) void convert_f16_kernel(
    const float* __restrict__ src, _Float16* __restrict__ dst,
    _Float16* __restrict__ dstn, size_t n) {
  size_t i = (size_t)blockIdx.x * 256 + threadIdx.x;
  if (i < n) {
    float v = src[i];
    dst[i] = (_Float16)v;
    if (dstn) dstn[i] = (_Float16)(-v);
  }
}

// RMS-norm over complex magnitude + magnitude (f16) for the gate GEMM.
__global__ __launch_bounds__(256) void norm_kernel(
    const float* __restrict__ xr, const float* __restrict__ xi,
    const float* __restrict__ normw,
    float* __restrict__ nr, float* __restrict__ ni, _Float16* __restrict__ magh) {
  const int token = blockIdx.x;          // 0..BS-1
  const int t = threadIdx.x;             // 0..255, 4 channels each
  const float* pr = xr + (size_t)token * Dn;
  const float* pi = xi + (size_t)token * Dn;
  float vr[4], vi[4];
  float ss = 0.f;
#pragma unroll
  for (int j = 0; j < 4; ++j) {
    int d = t + j * 256;
    vr[j] = pr[d]; vi[j] = pi[d];
    ss += vr[j] * vr[j] + vi[j] * vi[j];
  }
  __shared__ float red[256];
  red[t] = ss; __syncthreads();
  for (int o = 128; o > 0; o >>= 1) {
    if (t < o) red[t] += red[t + o];
    __syncthreads();
  }
  const float inv = rsqrtf(red[0] * (1.0f / (float)Dn) + 1e-6f);
#pragma unroll
  for (int j = 0; j < 4; ++j) {
    int d = t + j * 256;
    float w = normw[d];
    float a = vr[j] * inv * w;
    float b = vi[j] * inv * w;
    size_t o = (size_t)token * Dn + d;
    nr[o] = a; ni[o] = b;
    magh[o] = (_Float16)sqrtf(a * a + b * b + 1e-8f);
  }
}

// gated rotation
__global__ __launch_bounds__(256) void rotate_kernel(
    const float* __restrict__ nr, const float* __restrict__ ni,
    const float* __restrict__ gate, const float* __restrict__ theta,
    float* __restrict__ ur, float* __restrict__ ui) {
  size_t idx = (size_t)blockIdx.x * 256 + threadIdx.x;   // BSD
  int d = (int)(idx & (Dn - 1));
  float th = theta[d >> 3];                              // repeat(theta, 8)
  float st, ct; __sincosf(th, &st, &ct);
  float g = gate[idx];
  float a = nr[idx], b = ni[idx];
  ur[idx] = g * (a * ct - b * st);
  ui[idx] = g * (a * st + b * ct);
}

// causal depthwise (group=2ch) conv over time, K=4, + radial activation
__global__ __launch_bounds__(256) void conv_act_kernel(
    const float* __restrict__ ur, const float* __restrict__ ui,
    const float* __restrict__ convw, const float* __restrict__ convb,
    const float* __restrict__ act_thr,
    float* __restrict__ cr, float* __restrict__ ci) {
  size_t idx = (size_t)blockIdx.x * 256 + threadIdx.x;   // BSD
  int d = (int)(idx & (Dn - 1));
  size_t token = idx >> 10;
  int s = (int)(token & (Sn - 1));
  const float* wv = convw + (size_t)(2 * d) * 8;  // rows 2d,2d+1 -> 16 floats
  float accr = convb[2 * d];
  float acci = convb[2 * d + 1];
#pragma unroll
  for (int k = 0; k < 4; ++k) {
    int sp = s - 3 + k;
    if (sp >= 0) {
      size_t o = ((token - (size_t)s + sp) << 10) + d;
      float a = ur[o], b = ui[o];
      accr += wv[k] * a + wv[4 + k] * b;        // out ch 2d
      acci += wv[8 + k] * a + wv[12 + k] * b;   // out ch 2d+1
    }
  }
  float ea = __expf(act_thr[0]);
  float act = 1.f - __expf(-(accr * accr + acci * acci) * ea);
  cr[idx] = accr * act;
  ci[idx] = acci * act;
}

// dt projection: one wave32 per (token, g); 256-wide dot, wave reduce.
__global__ __launch_bounds__(256) void dt_kernel(
    const float* __restrict__ cr, const float* __restrict__ ci,
    const float* __restrict__ dtW, const float* __restrict__ dtb,
    float* __restrict__ dtm, float* __restrict__ dtp) {
  int gid  = blockIdx.x * 8 + (threadIdx.x >> 5);   // 0..BSG-1
  int lane = threadIdx.x & 31;
  size_t token = (size_t)(gid >> 3);
  int g = gid & 7;
  size_t base = token * Dn + (size_t)g * 128 + lane * 4;
  float a0 = 0.f, a1 = 0.f;
#pragma unroll
  for (int j = 0; j < 4; ++j) {
    float v = cr[base + j];
    int jj = lane * 4 + j;
    a0 += v * dtW[jj];
    a1 += v * dtW[256 + jj];
  }
#pragma unroll
  for (int j = 0; j < 4; ++j) {
    float v = ci[base + j];
    int jj = 128 + lane * 4 + j;
    a0 += v * dtW[jj];
    a1 += v * dtW[256 + jj];
  }
  for (int m = 16; m > 0; m >>= 1) {
    a0 += __shfl_xor(a0, m, 32);
    a1 += __shfl_xor(a1, m, 32);
  }
  if (lane == 0) {
    dtm[gid] = fminf(fmaxf(__expf(a0 + dtb[0]), 1e-4f), 2.0f);
    dtp[gid] = fminf(fmaxf(__expf(a1 + dtb[1]), 1e-4f), 2.0f);
  }
}

// ---------------------------------------------------------------------------
// TDM: issue a 2-D tensor_load_to_lds (D# per CDNA5 ISA)
// moving a (rows x width) f16 tile (row stride ldg elements) into LDS.
// ---------------------------------------------------------------------------
#if CM3_HAS_TDM
static __device__ __forceinline__ void tdm_load_2d(
    _Float16* lds_dst, const _Float16* gsrc, int width, int rows, int ldg) {
  // Generic LDS address: low 32 bits are the LDS byte offset.
  unsigned int lds_off = (unsigned int)(uintptr_t)lds_dst;
  unsigned long long ga = (unsigned long long)(uintptr_t)gsrc;
  v4u g0;
  g0.x = 1u;                                             // count=1, user D#
  g0.y = lds_off;                                        // lds_addr
  g0.z = (unsigned int)(ga & 0xFFFFFFFFull);             // global_addr[31:0]
  g0.w = (unsigned int)((ga >> 32) & 0x01FFFFFFull) | (2u << 30); // [56:32]|type=2
  unsigned long long td0 = (unsigned int)width;          // tensor_dim0
  unsigned long long td1 = (unsigned int)rows;           // tensor_dim1
  unsigned long long st0 = (unsigned int)ldg;            // tensor_dim0_stride
  unsigned long long q0 = 0x10000ull                     // data_size=1 (2 bytes)
                        | (td0 << 48);                   // tensor_dim0[15:0]
  unsigned long long q1 = (td0 >> 16)                    // tensor_dim0[31:16]
                        | (td1 << 16)                    // tensor_dim1
                        | ((unsigned long long)(width & 0xFFFF) << 48); // tile_dim0
  unsigned long long q2 = (unsigned long long)(rows & 0xFFFF)           // tile_dim1
                        | ((st0 & 0xFFFFFFFFull) << 32); // stride0[31:0]
  unsigned long long q3 = (st0 >> 32) & 0xFFFFull;       // stride0[47:32], stride1=0
  v8i g1;
  g1[0] = (int)(unsigned int)q0; g1[1] = (int)(unsigned int)(q0 >> 32);
  g1[2] = (int)(unsigned int)q1; g1[3] = (int)(unsigned int)(q1 >> 32);
  g1[4] = (int)(unsigned int)q2; g1[5] = (int)(unsigned int)(q2 >> 32);
  g1[6] = (int)(unsigned int)q3; g1[7] = (int)(unsigned int)(q3 >> 32);
  v4i z4 = {0, 0, 0, 0};
#if defined(__clang_major__) && (__clang_major__ >= 23)
  v8i z8 = {0, 0, 0, 0, 0, 0, 0, 0};
  __builtin_amdgcn_tensor_load_to_lds(g0, g1, z4, z4, z8, 0);
#else
  __builtin_amdgcn_tensor_load_to_lds(g0, g1, z4, z4, 0);
#endif
}
#endif

// ---------------------------------------------------------------------------
// WMMA GEMM engine: block tile 128(M) x 64(N), 8 waves (2x2 wmma each),
// K step 32 (v_wmma_f32_16x16x32_f16). A (MxK row-major) streamed from
// global; Bw (NxK row-major, ldb==K contiguous) staged once per block into
// LDS by the TDM, fragments then read via ds_load_b128.
// ---------------------------------------------------------------------------
static __device__ __forceinline__ v16h load_frag(
    const _Float16* __restrict__ base, int ld, int r0, int k0, int lane) {
  const int r  = r0 + (lane & 15);
  const int kb = k0 + ((lane >> 4) << 3);
  const _Float16* p = base + (size_t)r * ld + kb;
  v8h lo = *(const v8h*)(p);
  v8h hi = *(const v8h*)(p + 16);
  v16h f;
#pragma unroll
  for (int i = 0; i < 8; ++i) { f[i] = lo[i]; f[i + 8] = hi[i]; }
  return f;
}

enum { EPI_GATE = 0, EPI_ROWSCALE = 1, EPI_F16PAIR = 2, EPI_OUT = 3 };

template <bool DUAL, int EPI>
__global__ __launch_bounds__(256) void gemm_wmma(
    const _Float16* __restrict__ A1, const _Float16* __restrict__ A2,
    const _Float16* __restrict__ B1, const _Float16* __restrict__ B2,
    int K, int lda, int ldb, int ldc, int aOffZ, int cColOffZ,
    const float* __restrict__ bias,                     // EPI_GATE
    const float* __restrict__ rowscale, int rsStride,   // EPI_ROWSCALE
    float* __restrict__ Cf,                             // f32 out
    _Float16* __restrict__ C16a, _Float16* __restrict__ C16b, // EPI_F16PAIR
    const float* __restrict__ colscale, const float* __restrict__ sscale,
    const float* __restrict__ resid, float* __restrict__ Cout2) { // EPI_OUT
  constexpr int BM = 128, BN = 64;
  extern __shared__ _Float16 smem[];            // (DUAL?2:1) * BN * K halves
  const int z    = blockIdx.z;
  const int lane = threadIdx.x & 31;
  const int wave = threadIdx.x >> 5;
  const int wm = wave >> 1, wn = wave & 1;
  const int rowW = blockIdx.y * BM + wm * 32;   // wave M base (global)
  const int bnL  = wn * 32;                     // wave N base within block tile
  const int nT0  = blockIdx.x * BN;             // block's B-row base (global)
  const size_t tileElems = (size_t)BN * K;
  const _Float16* pA1 = A1 + (size_t)z * aOffZ;
  const _Float16* pA2 = DUAL ? (A2 + (size_t)z * aOffZ) : nullptr;

  // ---- stage B tile(s) into LDS ----
#if CM3_HAS_TDM
  if (threadIdx.x < 32) {                       // wave 0 drives the TDM
    tdm_load_2d(smem, B1 + (size_t)nT0 * ldb, K, BN, ldb);
    if (DUAL) tdm_load_2d(smem + tileElems, B2 + (size_t)nT0 * ldb, K, BN, ldb);
    __builtin_amdgcn_s_wait_tensorcnt(0);
  }
#else
  {
    const int mats = DUAL ? 2 : 1;
    const int kc = K >> 3;                      // v8h chunks per row
    const size_t total = (size_t)mats * BN * kc;
    for (size_t c = threadIdx.x; c < total; c += 256) {
      int mat = (int)(c / ((size_t)BN * kc));
      size_t rr = c - (size_t)mat * BN * kc;
      int r = (int)(rr / kc);
      int kk = (int)(rr % kc) << 3;
      const _Float16* src = (mat ? B2 : B1) + (size_t)(nT0 + r) * ldb + kk;
      *(v8h*)(smem + (size_t)mat * tileElems + (size_t)r * K + kk) =
          *(const v8h*)src;
    }
  }
#endif
  __syncthreads();

  v8f acc[2][2] = {};
  for (int k = 0; k < K; k += 32) {
    // CDNA5 prefetch for next K step (global_prefetch_b8)
    if (k + 32 < K)
      __builtin_prefetch(pA1 + (size_t)(rowW + (lane & 15)) * lda + k + 32, 0, 1);
    v16h a1[2], b1[2];
#pragma unroll
    for (int i = 0; i < 2; ++i) a1[i] = load_frag(pA1, lda, rowW + i * 16, k, lane);
#pragma unroll
    for (int j = 0; j < 2; ++j) b1[j] = load_frag(smem, K, bnL + j * 16, k, lane);
#pragma unroll
    for (int i = 0; i < 2; ++i)
#pragma unroll
      for (int j = 0; j < 2; ++j)
        acc[i][j] = __builtin_amdgcn_wmma_f32_16x16x32_f16(
            false, a1[i], false, b1[j], (short)0, acc[i][j], false, false);
    if (DUAL) {
      v16h a2[2], b2[2];
#pragma unroll
      for (int i = 0; i < 2; ++i) a2[i] = load_frag(pA2, lda, rowW + i * 16, k, lane);
#pragma unroll
      for (int j = 0; j < 2; ++j)
        b2[j] = load_frag(smem + tileElems, K, bnL + j * 16, k, lane);
#pragma unroll
      for (int i = 0; i < 2; ++i)
#pragma unroll
        for (int j = 0; j < 2; ++j)
          acc[i][j] = __builtin_amdgcn_wmma_f32_16x16x32_f16(
              false, a2[i], false, b2[j], (short)0, acc[i][j], false, false);
    }
  }

  // Epilogue. C/D layout: VGPR v, lane l -> row M = v + 8*(l>=16), col N = l%16.
  const int rBase = rowW + 8 * (lane >> 4);
  const int nLoc  = lane & 15;
#pragma unroll
  for (int i = 0; i < 2; ++i)
#pragma unroll
    for (int j = 0; j < 2; ++j)
#pragma unroll
      for (int v = 0; v < 8; ++v) {
        const int row = rBase + i * 16 + v;
        const int nb  = nT0 + bnL + j * 16 + nLoc;  // B-row / bias index
        const int col = nb + z * cColOffZ;          // output column
        const float val = acc[i][j][v];
        const size_t off = (size_t)row * ldc + col;
        if (EPI == EPI_GATE) {
          float t = val + bias[nb];
          Cf[off] = 1.0f / (1.0f + __expf(-t));
        } else if (EPI == EPI_ROWSCALE) {
          Cf[off] = val * rowscale[(size_t)row * rsStride + z];
        } else if (EPI == EPI_F16PAIR) {
          C16a[off] = (_Float16)val;
          if (C16b) C16b[off] = (_Float16)(-val);
        } else {  // EPI_OUT: scale, residual, interleave (stride 2)
          float o = val * colscale[col] * sscale[0];
          Cout2[off * 2] = resid[off] + o;
        }
      }
}

// ---------------------------------------------------------------------------
// Chunked parallel scan: 16 chunks of 256 steps (aligned with the renorm
// period). Pass1 computes per-chunk (prod, partial), combine chains the 16
// chunk states (with renorm), pass2 replays with correct initial state.
// ---------------------------------------------------------------------------
static __device__ __forceinline__ void ssm_coef(
    float dm, float dp, float sp, float ph, float& ar, float& ai) {
  float am = __expf(-dm * sp);
  float sn, cs; __sincosf(dp * ph, &sn, &cs);
  ar = am * cs; ai = am * sn;
}

__global__ __launch_bounds__(256) void scan_pass1(
    const float* __restrict__ bxr, const float* __restrict__ bxi,
    const float* __restrict__ dtm, const float* __restrict__ dtp,
    const float* __restrict__ logA, const float* __restrict__ Aph,
    float* __restrict__ cs) {
  int tid = blockIdx.x * 256 + threadIdx.x;     // 32768
  int q = tid & 2047;                           // (b,g,n)
  int chunk = tid >> 11;                        // 0..15
  int n = q & 63, g = (q >> 6) & 7, b = q >> 9;
  float sp = log1pf(__expf(logA[g * Nn + n]));  // softplus
  float ph = Aph[g * Nn + n];
  float hr = 0.f, hi = 0.f, Pr = 1.f, Pi = 0.f;
  int s0 = chunk * 256;
  for (int i = 0; i < 256; ++i) {
    size_t tg = ((size_t)b * Sn + (s0 + i)) * Gn + g;
    float ar, ai; ssm_coef(dtm[tg], dtp[tg], sp, ph, ar, ai);
    size_t o = tg * Nn + n;
    float nhr = ar * hr - ai * hi + bxr[o];
    float nhi = ar * hi + ai * hr + bxi[o];
    hr = nhr; hi = nhi;
    float nPr = ar * Pr - ai * Pi;
    float nPi = ar * Pi + ai * Pr;
    Pr = nPr; Pi = nPi;
  }
  float* c = cs + ((size_t)chunk * 2048 + q) * 4;
  c[0] = Pr; c[1] = Pi; c[2] = hr; c[3] = hi;
}

__global__ __launch_bounds__(256) void scan_combine(
    const float* __restrict__ cs, float* __restrict__ h0) {
  int q = blockIdx.x * 256 + threadIdx.x;       // 2048
  float hr = 0.f, hi = 0.f;
  for (int c = 0; c < 16; ++c) {
    h0[((size_t)c * 2048 + q) * 2 + 0] = hr;
    h0[((size_t)c * 2048 + q) * 2 + 1] = hi;
    const float* st = cs + ((size_t)c * 2048 + q) * 4;
    float nr2 = st[0] * hr - st[1] * hi + st[2];
    float ni2 = st[0] * hi + st[1] * hr + st[3];
    float hn = sqrtf(nr2 * nr2 + ni2 * ni2 + 1e-8f);
    float sc = fminf(hn, 100.f) / hn;            // renorm at chunk end
    hr = nr2 * sc; hi = ni2 * sc;
  }
}

__global__ __launch_bounds__(256) void scan_pass2(
    const float* __restrict__ bxr, const float* __restrict__ bxi,
    const float* __restrict__ dtm, const float* __restrict__ dtp,
    const float* __restrict__ logA, const float* __restrict__ Aph,
    const float* __restrict__ h0,
    float* __restrict__ Hr, float* __restrict__ Hi) {
  int tid = blockIdx.x * 256 + threadIdx.x;
  int q = tid & 2047;
  int chunk = tid >> 11;
  int n = q & 63, g = (q >> 6) & 7, b = q >> 9;
  float sp = log1pf(__expf(logA[g * Nn + n]));
  float ph = Aph[g * Nn + n];
  const float* hh = h0 + ((size_t)chunk * 2048 + q) * 2;
  float hr = hh[0], hi = hh[1];
  int s0 = chunk * 256;
  for (int i = 0; i < 256; ++i) {
    size_t tg = ((size_t)b * Sn + (s0 + i)) * Gn + g;
    float ar, ai; ssm_coef(dtm[tg], dtp[tg], sp, ph, ar, ai);
    size_t o = tg * Nn + n;
    float nhr = ar * hr - ai * hi + bxr[o];
    float nhi = ar * hi + ai * hr + bxi[o];
    if (i == 255) {
      float hn = sqrtf(nhr * nhr + nhi * nhi + 1e-8f);
      float sc = fminf(hn, 100.f) / hn;
      nhr *= sc; nhi *= sc;
    }
    hr = nhr; hi = nhi;
    Hr[o] = hr; Hi[o] = hi;
  }
}

// ---------------------------------------------------------------------------
// Host launcher
// ---------------------------------------------------------------------------
extern "C" void kernel_launch(void* const* d_in, const int* in_sizes, int n_in,
                              void* d_out, int out_size, void* d_ws, size_t ws_size,
                              hipStream_t stream) {
  (void)in_sizes; (void)n_in; (void)out_size; (void)ws_size;
  const float* x_r      = (const float*)d_in[0];
  const float* x_i      = (const float*)d_in[1];
  const float* norm_w   = (const float*)d_in[2];
  const float* sg_theta = (const float*)d_in[3];
  const float* sg_Wg    = (const float*)d_in[4];
  const float* sg_bg    = (const float*)d_in[5];
  const float* conv_w   = (const float*)d_in[6];
  const float* conv_b   = (const float*)d_in[7];
  const float* log_A    = (const float*)d_in[8];
  const float* A_phase  = (const float*)d_in[9];
  const float* B_Wr     = (const float*)d_in[10];
  const float* B_Wi     = (const float*)d_in[11];
  const float* C_Wr     = (const float*)d_in[12];
  const float* C_Wi     = (const float*)d_in[13];
  const float* dt_W     = (const float*)d_in[14];
  const float* dt_b     = (const float*)d_in[15];
  const float* out_Wr   = (const float*)d_in[16];
  const float* out_Wi   = (const float*)d_in[17];
  const float* act_thr  = (const float*)d_in[18];
  const float* ssm_sc   = (const float*)d_in[19];
  const float* res_sc   = (const float*)d_in[20];
  float* out = (float*)d_out;

  // Workspace layout (liveness-aliased regions). Total ~411 MB.
  char* w = (char*)d_ws;
  constexpr size_t F32BSD = BSD * 4, F16BSD = BSD * 2;
  constexpr size_t F32BSGN = BSGN * 4, F16BSGN = BSGN * 2;
  // R1 (67MB): nr -> cr -> Hr+Hi
  float*    nr    = (float*)(w + 0);
  float*    cr    = nr;
  float*    Hr    = nr;
  float*    Hi    = (float*)(w + F32BSGN);
  // R2 (67MB): ni -> ci -> Hr_h+Hi_h+Hin_h
  float*    ni    = (float*)(w + F32BSD);
  float*    ci    = ni;
  _Float16* Hr_h  = (_Float16*)(w + F32BSD);
  _Float16* Hi_h  = (_Float16*)(w + F32BSD + F16BSGN);
  _Float16* Hin_h = (_Float16*)(w + F32BSD + 2 * F16BSGN);
  // R3 (33.5MB): mag_h -> cr_h
  _Float16* mag_h = (_Float16*)(w + 2 * F32BSD);
  _Float16* cr_h  = mag_h;
  // R4 (67MB): gate -> ci_h + cin_h
  float*    gate  = (float*)(w + 2 * F32BSD + F16BSD);
  _Float16* ci_h  = (_Float16*)(w + 2 * F32BSD + F16BSD);
  _Float16* cin_h = (_Float16*)(w + 2 * F32BSD + 2 * F16BSD);
  // R5 (67MB): ur -> bxr + bxi
  float*    ur    = (float*)(w + 3 * F32BSD + F16BSD);
  float*    bxr   = ur;
  float*    bxi   = (float*)(w + 3 * F32BSD + F16BSD + F32BSGN);
  // R6 (67MB): ui -> yr_h + yi_h
  float*    ui    = (float*)(w + 4 * F32BSD + F16BSD);
  _Float16* yr_h  = (_Float16*)(w + 4 * F32BSD + F16BSD);
  _Float16* yi_h  = (_Float16*)(w + 4 * F32BSD + 2 * F16BSD);
  // R7 (33.5MB): yin_h
  _Float16* yin_h = (_Float16*)(w + 5 * F32BSD + F16BSD);
  // misc
  size_t mo = 5 * F32BSD + 2 * F16BSD;
  float*    dtm   = (float*)(w + mo); mo += BSG * 4;
  float*    dtp   = (float*)(w + mo); mo += BSG * 4;
  _Float16* Wg_h  = (_Float16*)(w + mo); mo += (size_t)Dn * Dn * 2;
  _Float16* Wr_h  = (_Float16*)(w + mo); mo += (size_t)Dn * Dn * 2;
  _Float16* Wi_h  = (_Float16*)(w + mo); mo += (size_t)Dn * Dn * 2;
  _Float16* BWr_h = (_Float16*)(w + mo); mo += (size_t)Nn * 128 * 2;
  _Float16* BWi_h = (_Float16*)(w + mo); mo += (size_t)Nn * 128 * 2;
  _Float16* CWr_h = (_Float16*)(w + mo); mo += (size_t)128 * Nn * 2;
  _Float16* CWi_h = (_Float16*)(w + mo); mo += (size_t)128 * Nn * 2;
  float*    csbuf = (float*)(w + mo); mo += (size_t)32768 * 4 * 4;
  float*    h0buf = (float*)(w + mo); mo += (size_t)32768 * 2 * 4;

  const dim3 blk(256);
  auto cvt = [&](const float* s, _Float16* d, _Float16* dn, size_t n) {
    convert_f16_kernel<<<dim3((unsigned)((n + 255) / 256)), blk, 0, stream>>>(s, d, dn, n);
  };

  // LDS bytes for each GEMM's staged B tile(s): (dual?2:1) * 64 rows * K * 2B
  const size_t sh_gate = (size_t)1 * 64 * Dn * 2;   // 128 KB
  const size_t sh_out  = (size_t)2 * 64 * Dn * 2;   // 256 KB (<= 320 KB WGP LDS)
  const size_t sh_bx   = (size_t)2 * 64 * 128 * 2;  //  32 KB
  const size_t sh_y    = (size_t)2 * 64 * Nn * 2;   //  16 KB

  // Weight conversions to f16
  cvt(sg_Wg,  Wg_h,  nullptr, (size_t)Dn * Dn);
  cvt(out_Wr, Wr_h,  nullptr, (size_t)Dn * Dn);
  cvt(out_Wi, Wi_h,  nullptr, (size_t)Dn * Dn);
  cvt(B_Wr,   BWr_h, nullptr, (size_t)Nn * 128);
  cvt(B_Wi,   BWi_h, nullptr, (size_t)Nn * 128);
  cvt(C_Wr,   CWr_h, nullptr, (size_t)128 * Nn);
  cvt(C_Wi,   CWi_h, nullptr, (size_t)128 * Nn);

  // 1) norm + magnitude
  norm_kernel<<<dim3(BS), blk, 0, stream>>>(x_r, x_i, norm_w, nr, ni, mag_h);

  // 2) gate GEMM: sigmoid(mag @ Wg^T + bg)   [M=16384, N=1024, K=1024]
  gemm_wmma<false, EPI_GATE><<<dim3(16, 128, 1), blk, sh_gate, stream>>>(
      mag_h, nullptr, Wg_h, nullptr, Dn, Dn, Dn, Dn, 0, 0,
      sg_bg, nullptr, 0, gate, nullptr, nullptr, nullptr, nullptr, nullptr, nullptr);

  // 3) gated rotation
  rotate_kernel<<<dim3((unsigned)(BSD / 256)), blk, 0, stream>>>(nr, ni, gate, sg_theta, ur, ui);

  // 4) causal conv + activation
  conv_act_kernel<<<dim3((unsigned)(BSD / 256)), blk, 0, stream>>>(
      ur, ui, conv_w, conv_b, act_thr, cr, ci);

  // 5) dt projection
  dt_kernel<<<dim3((unsigned)(BSG / 8)), blk, 0, stream>>>(cr, ci, dt_W, dt_b, dtm, dtp);

  // 6) activations -> f16 (with negated copy of ci for the real part)
  cvt(cr, cr_h, nullptr, BSD);
  cvt(ci, ci_h, cin_h, BSD);

  // 7) complex B-projection, dt_mag fused:  [M=16384, N=64, K=128] x G
  gemm_wmma<true, EPI_ROWSCALE><<<dim3(1, 128, Gn), blk, sh_bx, stream>>>(
      cr_h, cin_h, BWr_h, BWi_h, 128, Dn, 128, Gn * Nn, 128, Nn,
      nullptr, dtm, Gn, bxr, nullptr, nullptr, nullptr, nullptr, nullptr, nullptr);
  gemm_wmma<true, EPI_ROWSCALE><<<dim3(1, 128, Gn), blk, sh_bx, stream>>>(
      cr_h, ci_h, BWi_h, BWr_h, 128, Dn, 128, Gn * Nn, 128, Nn,
      nullptr, dtm, Gn, bxi, nullptr, nullptr, nullptr, nullptr, nullptr, nullptr);

  // 8) chunked parallel scan (16 chunks of 256, aligned with renorm period)
  scan_pass1<<<dim3(128), blk, 0, stream>>>(bxr, bxi, dtm, dtp, log_A, A_phase, csbuf);
  scan_combine<<<dim3(8), blk, 0, stream>>>(csbuf, h0buf);
  scan_pass2<<<dim3(128), blk, 0, stream>>>(bxr, bxi, dtm, dtp, log_A, A_phase, h0buf, Hr, Hi);

  // 9) H -> f16 (with negated Hi for the real part of C-projection)
  cvt(Hr, Hr_h, nullptr, BSGN);
  cvt(Hi, Hi_h, Hin_h, BSGN);

  // 10) complex C-projection -> yr/yi in f16 (+neg yi):  [M=16384, N=128, K=64] x G
  gemm_wmma<true, EPI_F16PAIR><<<dim3(2, 128, Gn), blk, sh_y, stream>>>(
      Hr_h, Hin_h, CWr_h, CWi_h, Nn, Gn * Nn, Nn, Dn, Nn, 128,
      nullptr, nullptr, 0, nullptr, yr_h, nullptr, nullptr, nullptr, nullptr, nullptr);
  gemm_wmma<true, EPI_F16PAIR><<<dim3(2, 128, Gn), blk, sh_y, stream>>>(
      Hr_h, Hi_h, CWi_h, CWr_h, Nn, Gn * Nn, Nn, Dn, Nn, 128,
      nullptr, nullptr, 0, nullptr, yi_h, yin_h, nullptr, nullptr, nullptr, nullptr);

  // 11) complex out-projection, fused ssm_scale*res_scale + residual + interleave
  gemm_wmma<true, EPI_OUT><<<dim3(16, 128, 1), blk, sh_out, stream>>>(
      yr_h, yin_h, Wr_h, Wi_h, Dn, Dn, Dn, Dn, 0, 0,
      nullptr, nullptr, 0, nullptr, nullptr, nullptr, ssm_sc, res_sc, x_r, out + 0);
  gemm_wmma<true, EPI_OUT><<<dim3(16, 128, 1), blk, sh_out, stream>>>(
      yr_h, yi_h, Wi_h, Wr_h, Dn, Dn, Dn, Dn, 0, 0,
      nullptr, nullptr, 0, nullptr, nullptr, nullptr, ssm_sc, res_sc, x_i, out + 1);
}
